// PCGNN_5342939316747
// MI455X (gfx1250) — matrix operand order repaired and checked by reading, not compiled, not run on hardware
//
#include <hip/hip_runtime.h>
#include <hip/hip_bf16.h>
#include <math.h>

typedef __attribute__((ext_vector_type(16))) _Float16 v16h;
typedef __attribute__((ext_vector_type(8)))  _Float16 v8h;
typedef __attribute__((ext_vector_type(8)))  float    v8f;
typedef __attribute__((ext_vector_type(4)))  int      v4i;

#define N_NODES 50000
#define N_EDGES 800000
#define IN_DIM  256
#define HIDDEN  128

// gfx1250 async Global->LDS copies (ASYNCcnt path), if the toolchain has them.
#if defined(__has_builtin)
#if __has_builtin(__builtin_amdgcn_global_load_async_to_lds_b128) && \
    __has_builtin(__builtin_amdgcn_s_wait_asynccnt)
#define PCGNN_ASYNC_LDS 1
#endif
#endif

#ifdef PCGNN_ASYNC_LDS
typedef __attribute__((address_space(1))) v4i gas_v4i;
typedef __attribute__((address_space(3))) v4i las_v4i;
// 16B per lane, global -> LDS, no VGPR round-trip.
static __device__ __forceinline__ void async_copy16(const _Float16* g, _Float16* l) {
    __builtin_amdgcn_global_load_async_to_lds_b128((gas_v4i*)g, (las_v4i*)l, 0, 0);
}
#endif

// ---------------------------------------------------------------------------
// A-matrix fragment loader (16x32 f16 tile, ISA 7.12.2 layout):
// lane<16 : row M=lane,   halves = K {kb..kb+7, 16+kb..16+kb+7}, kb=0
// lane>=16: row M=lane-16, same with kb=8
// Source is row-major f16 in LDS; two 16B chunks -> two ds_load_b128.
// ---------------------------------------------------------------------------
static __device__ __forceinline__ v16h load_a_frag(const _Float16* s, int stride, int kt) {
    int lane = threadIdx.x & 31;
    int row  = lane & 15;
    int kb   = (lane >> 4) << 3;          // 0 or 8
    const _Float16* p = s + row * stride + kt * 32 + kb;
    v8h lo = *(const v8h*)p;
    v8h hi = *(const v8h*)(p + 16);
    v16h a;
#pragma unroll
    for (int i = 0; i < 8; ++i) { a[i] = lo[i]; a[i + 8] = hi[i]; }
    return a;
}

// ---------------------------------------------------------------------------
// B-matrix fragment loader from pre-packed weights: fragment (kt,nt) stores
// 32 lanes x 16 halves contiguously -> two global_load_b128 per lane.
// ---------------------------------------------------------------------------
static __device__ __forceinline__ v16h load_b_frag(const _Float16* wpk, int nKt, int kt, int nt) {
    int lane = threadIdx.x & 31;
    const _Float16* p = wpk + ((size_t)((nt * nKt + kt) * 32 + lane) << 4);
    v8h lo = *(const v8h*)p;
    v8h hi = *(const v8h*)(p + 8);
    v16h b;
#pragma unroll
    for (int i = 0; i < 8; ++i) { b[i] = lo[i]; b[i + 8] = hi[i]; }
    return b;
}

// ---------------------------------------------------------------------------
// Pack f32 weight [K x HIDDEN] row-major into WMMA B-fragment order (f16).
// B 32x16 tile layout: lane = 16*h + col; halves i -> K = 8h + (i&7) + 16*(i>>3).
// ---------------------------------------------------------------------------
__global__ void pcgnn_pack_b(const float* __restrict__ W, _Float16* __restrict__ out, int K) {
    int tid = blockIdx.x * blockDim.x + threadIdx.x;
    int total = K * HIDDEN;
    if (tid >= total) return;
    int i    = tid & 15;
    int lane = (tid >> 4) & 31;
    int f    = tid >> 9;
    int nKt  = K >> 5;
    int kt   = f % nKt;
    int nt   = f / nKt;
    int k = kt * 32 + ((lane >> 4) << 3) + (i & 7) + ((i >> 3) << 4);
    int n = nt * 16 + (lane & 15);
    out[tid] = (_Float16)W[k * HIDDEN + n];
}

// ---------------------------------------------------------------------------
// Node GEMM: out = act(X @ W + b), X is [nrows x KDIM] (f32 or f16 selected
// at compile time), W packed f16, HIDDEN=128 outputs. 4 waves/block, 16 rows
// per wave. KDIM is a template constant so the K loop fully unrolls and the
// scheduler can pipeline global/ds loads against v_wmma with partial waits.
// ---------------------------------------------------------------------------
template <int KDIM, bool SRC32, bool RELU, bool W32, bool W16>
__global__ __launch_bounds__(128) void pcgnn_node_gemm(
    const float* __restrict__ X32, const _Float16* __restrict__ X16,
    const _Float16* __restrict__ Wpk, const float* __restrict__ bias,
    float* __restrict__ out32, _Float16* __restrict__ out16, int nrows)
{
    __shared__ __align__(16) _Float16 sX[4][16 * KDIM];
    constexpr int NKT  = KDIM / 32;
    constexpr int HALF = KDIM / 2;
    int wave = threadIdx.x >> 5, lane = threadIdx.x & 31;
    int r0 = (blockIdx.x * 4 + wave) * 16;

    // Stage 16 rows (f16) into LDS; each lane handles half of one row.
    int e = lane >> 1, hs = lane & 1;
    int rr = r0 + e; if (rr >= nrows) rr = nrows - 1;
    _Float16* dsts = &sX[wave][e * KDIM + hs * HALF];
    if (SRC32) {
        const float4* g = (const float4*)(X32 + (size_t)rr * KDIM + hs * HALF);
#pragma unroll
        for (int c = 0; c < HALF / 4; ++c) {
            float4 v = g[c];
            dsts[c * 4 + 0] = (_Float16)v.x; dsts[c * 4 + 1] = (_Float16)v.y;
            dsts[c * 4 + 2] = (_Float16)v.z; dsts[c * 4 + 3] = (_Float16)v.w;
        }
    } else {
        const _Float16* g16 = X16 + (size_t)rr * KDIM + hs * HALF;
#ifdef PCGNN_ASYNC_LDS
#pragma unroll
        for (int c = 0; c < HALF / 8; ++c) async_copy16(g16 + c * 8, dsts + c * 8);
        __builtin_amdgcn_s_wait_asynccnt(0);
#else
        const v8h* g = (const v8h*)g16;
        v8h* d = (v8h*)dsts;
#pragma unroll
        for (int c = 0; c < HALF / 8; ++c) d[c] = g[c];
#endif
    }
    __syncthreads();

    int col = lane & 15;
    int mo  = (lane >> 4) << 3;
    const _Float16* sa = &sX[wave][0];
#pragma unroll
    for (int nt = 0; nt < 8; ++nt) {
        v8f c;
        float bv = bias[nt * 16 + col];
#pragma unroll
        for (int r = 0; r < 8; ++r) c[r] = bv;
#pragma unroll
        for (int kt = 0; kt < NKT; ++kt) {
            v16h a = load_a_frag(sa, KDIM, kt);
            v16h b = load_b_frag(Wpk, NKT, kt, nt);
            c = __builtin_amdgcn_wmma_f32_16x16x32_f16(false, a, false, b,
                                                       (short)0, c, false, false);
        }
#pragma unroll
        for (int r = 0; r < 8; ++r) {
            int row = r0 + r + mo;
            if (row < nrows) {
                float v = c[r];
                if (RELU) v = fmaxf(v, 0.f);
                size_t o = (size_t)row * HIDDEN + nt * 16 + col;
                if (W32) out32[o] = v;
                if (W16) out16[o] = (_Float16)v;
            }
        }
    }
}

// ---------------------------------------------------------------------------
// Edge layer: per wave, 16 edges. Gather h16[src]/h16[dst] to LDS (async
// Global->LDS when available), WMMA the edge MLP (16x256 @ 256x128), fused
// ReLU + 128->1 dot + sigmoid, then msg = pick * t[src] scatter-added into
// acc[dst] with f32 atomics.
// ---------------------------------------------------------------------------
__global__ __launch_bounds__(128) void pcgnn_edge_layer(
    const long long* __restrict__ edges, const _Float16* __restrict__ h16,
    const float* __restrict__ t32,
    const _Float16* __restrict__ paW, const float* __restrict__ pa_b,
    const float* __restrict__ pb_w, const float* __restrict__ pb_b,
    float* __restrict__ acc)
{
    __shared__ __align__(16) _Float16 sXi[4][16 * HIDDEN];
    __shared__ __align__(16) _Float16 sXj[4][16 * HIDDEN];
    __shared__ float spick[4][16];
    int wave = threadIdx.x >> 5, lane = threadIdx.x & 31;
    int e0 = (blockIdx.x * 4 + wave) * 16;
    const long long* srcI = edges;
    const long long* dstI = edges + N_EDGES;

    // Prefetch upcoming edge indices (global_prefetch_b8).
    __builtin_prefetch(srcI + ((e0 + 1024 < N_EDGES) ? e0 + 1024 : 0), 0, 1);

    // Gather: lane pair per edge row; 16B chunks into LDS.
    int e = lane >> 1, hs = lane & 1;
    int eg = e0 + e; if (eg >= N_EDGES) eg = N_EDGES - 1;
    long long sn = srcI[eg], dn = dstI[eg];
    {
        const _Float16* gj = h16 + (size_t)sn * HIDDEN + hs * 64;
        const _Float16* gi = h16 + (size_t)dn * HIDDEN + hs * 64;
        _Float16* lj = &sXj[wave][e * HIDDEN + hs * 64];
        _Float16* li = &sXi[wave][e * HIDDEN + hs * 64];
#ifdef PCGNN_ASYNC_LDS
#pragma unroll
        for (int c = 0; c < 8; ++c) {
            async_copy16(gj + c * 8, lj + c * 8);
            async_copy16(gi + c * 8, li + c * 8);
        }
        __builtin_amdgcn_s_wait_asynccnt(0);
#else
        const v8h* vgj = (const v8h*)gj;
        const v8h* vgi = (const v8h*)gi;
        v8h* vlj = (v8h*)lj;
        v8h* vli = (v8h*)li;
#pragma unroll
        for (int c = 0; c < 8; ++c) { vlj[c] = vgj[c]; vli[c] = vgi[c]; }
#endif
    }
    __syncthreads();

    int col = lane & 15;
    v8f pd = {};
    for (int nt = 0; nt < 8; ++nt) {
        v8f c;
        float bv = pa_b[nt * 16 + col];
#pragma unroll
        for (int r = 0; r < 8; ++r) c[r] = bv;
#pragma unroll
        for (int kt = 0; kt < 8; ++kt) {
            const _Float16* sa = (kt < 4) ? &sXi[wave][0] : &sXj[wave][0];
            v16h a = load_a_frag(sa, HIDDEN, kt & 3);
            v16h b = load_b_frag(paW, 8, kt, nt);
            c = __builtin_amdgcn_wmma_f32_16x16x32_f16(false, a, false, b,
                                                       (short)0, c, false, false);
        }
        // ReLU + partial dot with pb_w (this lane's column slice).
        float w = pb_w[nt * 16 + col];
#pragma unroll
        for (int r = 0; r < 8; ++r) {
            float v = fmaxf(c[r], 0.f);
            pd[r] += v * w;
        }
    }
    // Reduce across the 16 lanes of each half (xor masks stay in-group).
#pragma unroll
    for (int m = 1; m < 16; m <<= 1) {
#pragma unroll
        for (int r = 0; r < 8; ++r) pd[r] += __shfl_xor(pd[r], m, 32);
    }
    if (col == 0) {
        float bb = pb_b[0];
        int base = (lane >> 4) << 3;
#pragma unroll
        for (int r = 0; r < 8; ++r) {
            float z = pd[r] + bb;
            spick[wave][base + r] = 1.f / (1.f + __expf(-z));
        }
    }
    __syncthreads();

    // Scatter: msg = pick * t[src]; 32 lanes x 4 consecutive floats.
    for (int e2 = 0; e2 < 16; ++e2) {
        int ee = e0 + e2;
        if (ee >= N_EDGES) break;
        long long s2 = srcI[ee], d2 = dstI[ee];
        float p = spick[wave][e2];
        float4 tv = *(const float4*)(t32 + (size_t)s2 * HIDDEN + lane * 4);
        float* dp = acc + (size_t)d2 * HIDDEN + lane * 4;
        __hip_atomic_fetch_add(dp + 0, p * tv.x, __ATOMIC_RELAXED, __HIP_MEMORY_SCOPE_AGENT);
        __hip_atomic_fetch_add(dp + 1, p * tv.y, __ATOMIC_RELAXED, __HIP_MEMORY_SCOPE_AGENT);
        __hip_atomic_fetch_add(dp + 2, p * tv.z, __ATOMIC_RELAXED, __HIP_MEMORY_SCOPE_AGENT);
        __hip_atomic_fetch_add(dp + 3, p * tv.w, __ATOMIC_RELAXED, __HIP_MEMORY_SCOPE_AGENT);
    }
}

// ReLU + f32 -> f16 conversion of node features.
__global__ void pcgnn_relu_h16(const float* __restrict__ acc, _Float16* __restrict__ h16, int n) {
    int i = blockIdx.x * blockDim.x + threadIdx.x;
    if (i < n) h16[i] = (_Float16)fmaxf(acc[i], 0.f);
}

// Final classifier: [N x 128] @ [128 x 2] + b  (tiny, VALU).
__global__ void pcgnn_classifier(const float* __restrict__ acc, const float* __restrict__ cw,
                                 const float* __restrict__ cb, float* __restrict__ out) {
    int t = blockIdx.x * blockDim.x + threadIdx.x;
    if (t >= N_NODES * 2) return;
    int n = t >> 1, o = t & 1;
    const float* row = acc + (size_t)n * HIDDEN;
    float s = cb[o];
#pragma unroll 8
    for (int c = 0; c < HIDDEN; ++c) s += row[c] * cw[c * 2 + o];
    out[t] = s;
}

extern "C" void kernel_launch(void* const* d_in, const int* in_sizes, int n_in,
                              void* d_out, int out_size, void* d_ws, size_t ws_size,
                              hipStream_t stream) {
    (void)in_sizes; (void)n_in; (void)out_size; (void)ws_size;
    const float*     x       = (const float*)d_in[0];
    const long long* edges   = (const long long*)d_in[1];
    const float*     enc_w   = (const float*)d_in[2];
    const float*     enc_b   = (const float*)d_in[3];
    const float*     lin1_w  = (const float*)d_in[4];
    const float*     lin1_b  = (const float*)d_in[5];
    const float*     p1a_w   = (const float*)d_in[6];
    const float*     p1a_b   = (const float*)d_in[7];
    const float*     p1b_w   = (const float*)d_in[8];
    const float*     p1b_b   = (const float*)d_in[9];
    const float*     lin2_w  = (const float*)d_in[10];
    const float*     lin2_b  = (const float*)d_in[11];
    const float*     p2a_w   = (const float*)d_in[12];
    const float*     p2a_b   = (const float*)d_in[13];
    const float*     p2b_w   = (const float*)d_in[14];
    const float*     p2b_b   = (const float*)d_in[15];
    const float*     cls_w   = (const float*)d_in[16];
    const float*     cls_b   = (const float*)d_in[17];

    char* ws = (char*)d_ws;
    _Float16* h16    = (_Float16*)(ws + 0);            // N*128 f16   = 12.8 MB
    float*    t32    = (float*)(ws + 12800000);        // N*128 f32   = 25.6 MB
    float*    acc    = (float*)(ws + 38400000);        // N*128 f32   = 25.6 MB
    _Float16* encWp  = (_Float16*)(ws + 64000000);     // 256*128 f16
    _Float16* lin1Wp = (_Float16*)(ws + 64065536);     // 128*128 f16
    _Float16* pa1Wp  = (_Float16*)(ws + 64098304);     // 256*128 f16
    _Float16* lin2Wp = (_Float16*)(ws + 64163840);     // 128*128 f16
    _Float16* pa2Wp  = (_Float16*)(ws + 64196608);     // 256*128 f16

    // Pack weights to WMMA B-fragment order.
    pcgnn_pack_b<<<(IN_DIM * HIDDEN + 255) / 256, 256, 0, stream>>>(enc_w, encWp, IN_DIM);
    pcgnn_pack_b<<<(HIDDEN * HIDDEN + 255) / 256, 256, 0, stream>>>(lin1_w, lin1Wp, HIDDEN);
    pcgnn_pack_b<<<(IN_DIM * HIDDEN + 255) / 256, 256, 0, stream>>>(p1a_w, pa1Wp, IN_DIM);
    pcgnn_pack_b<<<(HIDDEN * HIDDEN + 255) / 256, 256, 0, stream>>>(lin2_w, lin2Wp, HIDDEN);
    pcgnn_pack_b<<<(IN_DIM * HIDDEN + 255) / 256, 256, 0, stream>>>(p2a_w, pa2Wp, IN_DIM);

    int nblk = (N_NODES + 63) / 64;
    int eblk = (N_EDGES + 63) / 64;
    size_t accBytes = (size_t)N_NODES * HIDDEN * sizeof(float);

    // h = relu(x @ enc_w + enc_b), stored f16 for gathers.
    pcgnn_node_gemm<IN_DIM, true, true, false, true>
        <<<nblk, 128, 0, stream>>>(x, nullptr, encWp, enc_b, nullptr, h16, N_NODES);
    // t = h @ lin1_w + lin1_b  (per-node, hoisted out of edge loop)
    pcgnn_node_gemm<HIDDEN, false, false, true, false>
        <<<nblk, 128, 0, stream>>>(nullptr, h16, lin1Wp, lin1_b, t32, nullptr, N_NODES);
    (void)hipMemsetAsync(acc, 0, accBytes, stream);
    pcgnn_edge_layer<<<eblk, 128, 0, stream>>>(edges, h16, t32, pa1Wp, p1a_b,
                                               p1b_w, p1b_b, acc);
    // h = relu(layer1 out)
    pcgnn_relu_h16<<<(N_NODES * HIDDEN + 255) / 256, 256, 0, stream>>>(acc, h16, N_NODES * HIDDEN);
    pcgnn_node_gemm<HIDDEN, false, false, true, false>
        <<<nblk, 128, 0, stream>>>(nullptr, h16, lin2Wp, lin2_b, t32, nullptr, N_NODES);
    (void)hipMemsetAsync(acc, 0, accBytes, stream);
    pcgnn_edge_layer<<<eblk, 128, 0, stream>>>(edges, h16, t32, pa2Wp, p2a_b,
                                               p2b_w, p2b_b, acc);
    // out = h'' @ cls_w + cls_b (no relu before classifier)
    pcgnn_classifier<<<(N_NODES * 2 + 255) / 256, 256, 0, stream>>>(acc, cls_w, cls_b,
                                                                    (float*)d_out);
}
